// LiquidHead_31301721653287
// MI455X (gfx1250) — matrix operand order repaired and checked by reading, not compiled, not run on hardware
//
#include <hip/hip_runtime.h>
#include <hip/hip_bf16.h>
#include <math.h>

typedef __bf16 bf16_t;
typedef __attribute__((ext_vector_type(16))) __bf16 v16bf;
typedef __attribute__((ext_vector_type(8)))  __bf16 v8bf;
typedef __attribute__((ext_vector_type(8)))  float  v8f;

#define LNN_DT 0.05f

// ---------- fp32 -> bf16 streaming convert ----------
__global__ void k_cvt_bf16(const float* __restrict__ src, bf16_t* __restrict__ dst, size_t n) {
  size_t i = ((size_t)blockIdx.x * blockDim.x + threadIdx.x) * 4;
  if (i + 3 < n) {
    float4 v = *(const float4*)(src + i);
    dst[i + 0] = (bf16_t)v.x; dst[i + 1] = (bf16_t)v.y;
    dst[i + 2] = (bf16_t)v.z; dst[i + 3] = (bf16_t)v.w;
  } else {
    for (; i < n; ++i) dst[i] = (bf16_t)src[i];
  }
}

// ---------- rec_weights transpose (rT[j][i] = rec[i][j]) to bf16 ----------
__global__ void k_transpose_bf16(const float* __restrict__ src, bf16_t* __restrict__ dst, int H) {
  __shared__ float tile[32][33];
  const int bx = blockIdx.x, by = blockIdx.y;
  for (int i = threadIdx.x; i < 32 * 32; i += blockDim.x) {
    int r = i >> 5, c = i & 31;
    tile[r][c] = src[(size_t)(by * 32 + r) * H + bx * 32 + c];
  }
  __syncthreads();
  for (int i = threadIdx.x; i < 32 * 32; i += blockDim.x) {
    int r = i >> 5, c = i & 31;
    dst[(size_t)(bx * 32 + r) * H + by * 32 + c] = (bf16_t)tile[c][r];
  }
}

// ---------- per-launch init: h0(bf16) = 0, tau_softplus, barrier counter ----------
__global__ void k_init(bf16_t* __restrict__ h0, float* __restrict__ tsp,
                       const float* __restrict__ tau, unsigned* __restrict__ cnt,
                       int BH, int H) {
  for (int i = threadIdx.x; i < BH; i += blockDim.x) h0[i] = (bf16_t)0.f;
  for (int i = threadIdx.x; i < H; i += blockDim.x) {
    float t = tau[i];
    tsp[i] = (t > 20.f) ? t : log1pf(expf(t));
  }
  if (threadIdx.x == 0) *cnt = 0u;
}

// ---------- input projection: u[s][b][h] = x[b][s][:] . W_in[h][:] + b_in[h] ----------
// One 16x16 WMMA tile per wave: M = batch (B==16), tiles over (s, n).
__global__ void __launch_bounds__(256)
k_inproj(const bf16_t* __restrict__ xb, const bf16_t* __restrict__ wb,
         const float* __restrict__ bin, bf16_t* __restrict__ ub,
         int S, int D, int H) {
  const int lane = threadIdx.x & 31;
  const int wave = threadIdx.x >> 5;
  const int ntn  = H >> 4;
  const int gw   = blockIdx.x * 8 + wave;
  const int nt   = gw % ntn;
  const int s    = gw / ntn;
  if (s >= S) return;

  const int ln   = lane & 15;
  const int hi   = lane >> 4;      // K-half select
  const int sub  = hi * 8;
  const int koff = hi * 16;
  const int n0   = nt * 16;

  const bf16_t* xrow = xb + ((size_t)ln * S + s) * D;   // row b = ln
  const bf16_t* wrow = wb + (size_t)(n0 + ln) * D;      // row h = n0+ln (B operand col)

  v8f acc = {};
#pragma unroll 4
  for (int k0 = 0; k0 < D; k0 += 32) {
    v8bf a0 = *(const v8bf*)(xrow + k0 + koff);
    v8bf a1 = *(const v8bf*)(xrow + k0 + koff + 8);
    v16bf A = __builtin_shufflevector(a0, a1, 0,1,2,3,4,5,6,7,8,9,10,11,12,13,14,15);
    v16bf Bf = *(const v16bf*)(wrow + k0 + koff);
    acc = __builtin_amdgcn_wmma_f32_16x16x32_bf16(false, A, false, Bf, (short)0, acc, false, false);
  }
  const float bias = bin[n0 + ln];
#pragma unroll
  for (int v = 0; v < 8; ++v) {
    int m = v + hi * 8;                                  // batch row of C fragment
    ub[((size_t)s * 16 + m) * H + n0 + ln] = (bf16_t)(acc[v] + bias);
  }
}

// ---------- sequential liquid scan ----------
// 64 persistent WGs; R slice LDS-resident; fp32 state lives in wave0 registers;
// inter-step h published in bf16 (A-operand format consumers only).
__global__ void __launch_bounds__(128)
k_scan(const bf16_t* __restrict__ rT, const bf16_t* __restrict__ ub,
       const float* __restrict__ tsp, bf16_t* __restrict__ hA, bf16_t* __restrict__ hB,
       float* __restrict__ out, unsigned* __restrict__ cnt,
       int S, int H) {
  __shared__ __align__(16) bf16_t Rs[16 * 1024];  // 16 output cols x K=H, 32 KB
  __shared__ float red[3 * 32 * 8];               // partial-C from waves 1..3

  const int lane = threadIdx.x & 31;
  const int wave = threadIdx.x >> 5;
  const int ln = lane & 15, hi = lane >> 4;
  const int sub = hi * 8, koff = hi * 16;
  const int n0 = blockIdx.x * 16;

  // Stage recT rows n0..n0+15 into LDS (resident for all 2048 steps)
  for (int idx = threadIdx.x; idx < 16 * (H >> 3); idx += blockDim.x) {
    int r = idx / (H >> 3), c = idx % (H >> 3);
    *(v8bf*)&Rs[r * 1024 + c * 8] = *(const v8bf*)&rT[(size_t)(n0 + r) * H + c * 8];
  }
  __syncthreads();

  const int kPerWave = H >> 2;           // 4 waves split K
  const int kbase = wave * kPerWave;
  const float mytsp = tsp[n0 + ln];

  v8f hst = {};                          // fp32 state for this WG's tile (wave0 only)

  for (int t = 0; t < S; ++t) {
    const bf16_t* hin  = (t & 1) ? hB : hA;
    bf16_t*       hout = (t & 1) ? hA : hB;
    const bf16_t* hrow = hin + ln * H;

    v8f acc = {};
#pragma unroll 2
    for (int k0 = kbase; k0 < kbase + kPerWave; k0 += 32) {
      v8bf a0 = *(const v8bf*)(hrow + k0 + koff);
      v8bf a1 = *(const v8bf*)(hrow + k0 + koff + 8);
      v16bf A = __builtin_shufflevector(a0, a1, 0,1,2,3,4,5,6,7,8,9,10,11,12,13,14,15);
      v16bf Bf = *(const v16bf*)&Rs[ln * 1024 + k0 + koff];
      acc = __builtin_amdgcn_wmma_f32_16x16x32_bf16(false, A, false, Bf, (short)0, acc, false, false);
    }

    if (wave != 0) {
#pragma unroll
      for (int v = 0; v < 8; ++v) red[(wave - 1) * 256 + lane * 8 + v] = acc[v];
    }
    __syncthreads();

    if (wave == 0) {
#pragma unroll
      for (int w = 0; w < 3; ++w)
#pragma unroll
        for (int v = 0; v < 8; ++v) acc[v] += red[w * 256 + lane * 8 + v];

#pragma unroll
      for (int v = 0; v < 8; ++v) {
        const int m = v + hi * 8;
        const int col = n0 + ln;
        float u  = (float)ub[((size_t)t * 16 + m) * H + col];
        float f  = tanhf(acc[v] + u);
        float ho = hst[v];
        float hn = ho + (-ho / mytsp + f) * LNN_DT;
        hn = fminf(10.f, fmaxf(-10.f, hn));
        hst[v] = hn;
        hout[m * H + col] = (bf16_t)hn;
        if (t == S - 1) out[m * H + col] = hn;
        if (t + 1 < S) __builtin_prefetch(&ub[((size_t)(t + 1) * 16 + m) * H + col], 0, 1);
      }
    }
    __syncthreads();

    // device-wide step barrier (monotonic counter, no reset races)
    if (threadIdx.x == 0) {
      __threadfence();
      __hip_atomic_fetch_add(cnt, 1u, __ATOMIC_RELEASE, __HIP_MEMORY_SCOPE_AGENT);
      const unsigned target = (unsigned)gridDim.x * (unsigned)(t + 1);
      while (__hip_atomic_load(cnt, __ATOMIC_ACQUIRE, __HIP_MEMORY_SCOPE_AGENT) < target)
        __builtin_amdgcn_s_sleep(1);
    }
    __syncthreads();
  }
}

extern "C" void kernel_launch(void* const* d_in, const int* in_sizes, int n_in,
                              void* d_out, int out_size, void* d_ws, size_t ws_size,
                              hipStream_t stream) {
  const float* x   = (const float*)d_in[0];
  const float* Win = (const float*)d_in[1];
  const float* bin = (const float*)d_in[2];
  const float* rec = (const float*)d_in[3];
  const float* tau = (const float*)d_in[4];
  float* out = (float*)d_out;

  const int H = in_sizes[2];            // 1024
  const int D = in_sizes[1] / H;        // 1024
  const int B = out_size / H;           // 16 (== WMMA M)
  const int S = in_sizes[0] / (B * D);  // 2048

  char* ws = (char*)d_ws;
  size_t off = 0;
  auto take = [&](size_t bytes) -> char* {
    char* p = ws + off;
    off = (off + bytes + 255) & ~(size_t)255;
    return p;
  };
  bf16_t*  xb  = (bf16_t*)take((size_t)B * S * D * 2);   // 64 MB
  bf16_t*  wb  = (bf16_t*)take((size_t)H * D * 2);       //  2 MB
  bf16_t*  rT  = (bf16_t*)take((size_t)H * H * 2);       //  2 MB
  bf16_t*  ubf = (bf16_t*)take((size_t)S * B * H * 2);   // 64 MB
  float*   tsp = (float*)take((size_t)H * 4);
  bf16_t*  hA  = (bf16_t*)take((size_t)B * H * 2);       // 32 KB
  bf16_t*  hB  = (bf16_t*)take((size_t)B * H * 2);       // 32 KB
  unsigned* cnt = (unsigned*)take(256);

  const size_t nx = (size_t)B * S * D;
  k_cvt_bf16<<<(unsigned)((nx / 4 + 255) / 256), 256, 0, stream>>>(x, xb, nx);
  const size_t nw = (size_t)H * D;
  k_cvt_bf16<<<(unsigned)((nw / 4 + 255) / 256), 256, 0, stream>>>(Win, wb, nw);

  dim3 tg(H / 32, H / 32);
  k_transpose_bf16<<<tg, 256, 0, stream>>>(rec, rT, H);

  k_init<<<1, 256, 0, stream>>>(hA, tsp, tau, cnt, B * H, H);

  const int ntn = H / 16;                       // 64 N-tiles
  const int blocks = (S * ntn) / 8;             // 8 waves (tiles) per block
  k_inproj<<<blocks, 256, 0, stream>>>(xb, wb, bin, ubf, S, D, H);

  k_scan<<<ntn, 128, 0, stream>>>(rT, ubf, tsp, hA, hB, out, cnt, S, H);
}
